// TSGEncoder_77627238908074
// MI455X (gfx1250) — matrix operand order repaired and checked by reading, not compiled, not run on hardware
//
#include <hip/hip_runtime.h>

// ---- problem constants (match reference) ----
static constexpr int cB   = 16;
static constexpr int cS   = 256;
static constexpr int cD   = 768;
static constexpr int cH   = 12;
static constexpr int cDH  = 64;
static constexpr int cDEP = 40;
static constexpr int cDFF = 768;
static constexpr int cPOS = 30;
static constexpr int cL   = 2;
static constexpr int cNLOC= 2;
static constexpr int cWSZ = 3;
static constexpr int cVALF= 800;   // padded 768+30 -> 800 (val buffer lda)
static constexpr int cCATF= 1568;  // padded 1566 -> 1568 (cat buffer lda)
#define NEGV (-1e18f)

typedef __attribute__((ext_vector_type(16))) _Float16 v16h;
typedef __attribute__((ext_vector_type(8)))  float    v8f;

// ================= generic batched WMMA GEMM =================
// C = epilogue(A @ B) with flexible strides and batch-offset modes.
// B is either f32 (Bm, strides ldbK/ldbN) or pre-transposed f16 (B16, [N,Kpad],
// zero-padded in K to mult-of-32 and in N to mult-of-64 -> unguarded v16h loads).
// mode 0: offA=z*sAz, offB=z*sBz, offC=z*sCz          (B16 off = z*sBz)
// mode 1 (scores):  z=(b,h): offA=b*S*D+h*DH, offB16 same layout, offC=z*S*S
// mode 2 (ctx):     offA=z*S*S, offB16=z*DH*S (VT16), offC=b*S*D+h*DH
// mode 3 (sv-add):  offA=z*S*DEP, offB16=0, offC=b*S*D+h*DH
struct GemmP {
  const float* A; const float* Bm; const _Float16* B16;
  const float* bias; const float* resid;
  float* C;
  int M, N, K;
  int lda, ldbK, ldbN, ldb16, ldc;
  long long sAz, sBz, sCz;
  int mode, flags;     // flags: 1=relu, 2=accumulate into C
  float scale;
};
#define GF_RELU 1
#define GF_ACC  2

__device__ __forceinline__ v16h load_a_tile(const float* A, long long base, int lda,
                                            int m, int kk, int hsel, int M, int K) {
  v16h a;
  if (m < M && kk + 32 <= K) {
    const float* p = A + base + (long long)m * lda + kk + 16 * hsel;
    float4 x0 = *(const float4*)(p);
    float4 x1 = *(const float4*)(p + 4);
    float4 x2 = *(const float4*)(p + 8);
    float4 x3 = *(const float4*)(p + 12);
    a[0]=(_Float16)x0.x;  a[1]=(_Float16)x0.y;  a[2]=(_Float16)x0.z;  a[3]=(_Float16)x0.w;
    a[4]=(_Float16)x1.x;  a[5]=(_Float16)x1.y;  a[6]=(_Float16)x1.z;  a[7]=(_Float16)x1.w;
    a[8]=(_Float16)x2.x;  a[9]=(_Float16)x2.y;  a[10]=(_Float16)x2.z; a[11]=(_Float16)x2.w;
    a[12]=(_Float16)x3.x; a[13]=(_Float16)x3.y; a[14]=(_Float16)x3.z; a[15]=(_Float16)x3.w;
  } else {
#pragma unroll
    for (int j = 0; j < 16; ++j) {
      int k = kk + 16 * hsel + j;
      float v = (m < M && k < K) ? A[base + (long long)m * lda + k] : 0.f;
      a[j] = (_Float16)v;
    }
  }
  return a;
}

__device__ __forceinline__ v16h load_b_tile(const float* Bm, long long base, int ldbK, int ldbN,
                                            int n, int kk, int hsel, int N, int K) {
  v16h b;
  if (n < N && kk + 32 <= K) {
    if (ldbK == 1) {
      const float* p = Bm + base + (long long)n * ldbN + kk + 16 * hsel;
      float4 x0 = *(const float4*)(p);
      float4 x1 = *(const float4*)(p + 4);
      float4 x2 = *(const float4*)(p + 8);
      float4 x3 = *(const float4*)(p + 12);
      b[0]=(_Float16)x0.x;  b[1]=(_Float16)x0.y;  b[2]=(_Float16)x0.z;  b[3]=(_Float16)x0.w;
      b[4]=(_Float16)x1.x;  b[5]=(_Float16)x1.y;  b[6]=(_Float16)x1.z;  b[7]=(_Float16)x1.w;
      b[8]=(_Float16)x2.x;  b[9]=(_Float16)x2.y;  b[10]=(_Float16)x2.z; b[11]=(_Float16)x2.w;
      b[12]=(_Float16)x3.x; b[13]=(_Float16)x3.y; b[14]=(_Float16)x3.z; b[15]=(_Float16)x3.w;
    } else {
#pragma unroll
      for (int j = 0; j < 16; ++j) {
        int k = kk + 16 * hsel + j;
        b[j] = (_Float16)Bm[base + (long long)k * ldbK + (long long)n * ldbN];
      }
    }
  } else {
#pragma unroll
    for (int j = 0; j < 16; ++j) {
      int k = kk + 16 * hsel + j;
      float v = (n < N && k < K) ? Bm[base + (long long)k * ldbK + (long long)n * ldbN] : 0.f;
      b[j] = (_Float16)v;
    }
  }
  return b;
}

__global__ void __launch_bounds__(256) wmma_gemm_kernel(GemmP g) {
  const int z = blockIdx.z;
  long long offA, offB, offC;
  if (g.mode == 0) {
    offA = (long long)z * g.sAz; offB = (long long)z * g.sBz; offC = (long long)z * g.sCz;
  } else if (g.mode == 1) {
    int b = z / cH, hh = z % cH;
    offA = (long long)b * cS * cD + hh * cDH; offB = offA; offC = (long long)z * cS * cS;
  } else if (g.mode == 2) {
    int b = z / cH, hh = z % cH;
    offA = (long long)z * cS * cS; offB = (long long)z * cDH * cS;  // VT16 layout
    offC = (long long)b * cS * cD + hh * cDH;
  } else {
    int b = z / cH, hh = z % cH;
    offA = (long long)z * cS * cDEP; offB = 0; offC = (long long)b * cS * cD + hh * cDH;
  }
  const int wave = threadIdx.x >> 5;
  const int lane = threadIdx.x & 31;
  const int hsel = lane >> 4;
  const int r    = lane & 15;
  const int m0 = blockIdx.x * 64 + (wave >> 1) * 16;
  const int n0 = blockIdx.y * 64 + (wave & 1) * 32;
  if (m0 >= g.M || n0 >= g.N) return;     // wave-uniform: EXEC stays all-ones for WMMA

  v8f c0 = {}; v8f c1 = {};
  if (g.B16) {
    // pre-transposed/padded f16 B: one unguarded 32-byte v16h load per tile per step
    const _Float16* bp0 = g.B16 + offB + (long long)(n0 + r) * g.ldb16 + 16 * hsel;
    const _Float16* bp1 = bp0 + 16LL * g.ldb16;
    for (int kk = 0; kk < g.K; kk += 32) {
      v16h a  = load_a_tile(g.A, offA, g.lda, m0 + r, kk, hsel, g.M, g.K);
      v16h b0 = *(const v16h*)(bp0 + kk);
      v16h b1 = *(const v16h*)(bp1 + kk);
      c0 = __builtin_amdgcn_wmma_f32_16x16x32_f16(false, a, false, b0, (short)0, c0, false, false);
      c1 = __builtin_amdgcn_wmma_f32_16x16x32_f16(false, a, false, b1, (short)0, c1, false, false);
    }
  } else {
    for (int kk = 0; kk < g.K; kk += 32) {
      v16h a  = load_a_tile(g.A, offA, g.lda, m0 + r, kk, hsel, g.M, g.K);
      v16h b0 = load_b_tile(g.Bm, offB, g.ldbK, g.ldbN, n0 + r,      kk, hsel, g.N, g.K);
      v16h b1 = load_b_tile(g.Bm, offB, g.ldbK, g.ldbN, n0 + 16 + r, kk, hsel, g.N, g.K);
      c0 = __builtin_amdgcn_wmma_f32_16x16x32_f16(false, a, false, b0, (short)0, c0, false, false);
      c1 = __builtin_amdgcn_wmma_f32_16x16x32_f16(false, a, false, b1, (short)0, c1, false, false);
    }
  }
#pragma unroll
  for (int i = 0; i < 8; ++i) {
    int m = m0 + i + 8 * hsel;
    if (m >= g.M) continue;
#pragma unroll
    for (int tt = 0; tt < 2; ++tt) {
      int n = n0 + tt * 16 + r;
      if (n >= g.N) continue;
      float v = tt ? c1[i] : c0[i];
      if (g.bias) v += g.bias[n];
      v *= g.scale;
      if (g.flags & GF_RELU) v = fmaxf(v, 0.f);
      long long idx = offC + (long long)m * g.ldc + n;
      if (g.resid) v += g.resid[idx];
      if (g.flags & GF_ACC) v += g.C[idx];
      g.C[idx] = v;
    }
  }
}

// ===== weight convert+transpose: out[n*Kpad+k] = (k<K && n<N) ? in[k*sK+n*sN] : 0 =====
__global__ void __launch_bounds__(256) conv_wt_kernel(const float* in, _Float16* out,
                                                      int K, int N, int sK, int sN,
                                                      int Kpad, long long total) {
  long long i = (long long)blockIdx.x * 256 + threadIdx.x;
  if (i >= total) return;
  int k = (int)(i % Kpad);
  long long n = i / Kpad;
  float v = (k < K && n < N) ? in[(long long)k * sK + n * sN] : 0.f;
  out[i] = (_Float16)v;
}

// ===== straight f32 -> f16 cast copy (layout preserved) =====
__global__ void __launch_bounds__(256) conv_cast_kernel(const float* in, _Float16* out,
                                                        long long total) {
  long long i = (long long)blockIdx.x * 256 + threadIdx.x;
  if (i >= total) return;
  out[i] = (_Float16)in[i];
}

// ===== V [B,S,D] -> VT16 [(b*H+h), d, k] f16 (per-head transpose) =====
__global__ void __launch_bounds__(256) conv_vt_kernel(const float* V, _Float16* VT) {
  long long i = (long long)blockIdx.x * 256 + threadIdx.x;
  const long long total = (long long)cB * cH * cDH * cS;
  if (i >= total) return;
  int k = (int)(i & (cS - 1));
  int d = (int)((i >> 8) & (cDH - 1));
  int z = (int)(i >> 14);
  int b = z / cH, hh = z % cH;
  VT[i] = (_Float16)V[((long long)(b * cS + k)) * cD + hh * cDH + d];
}

// ============== scores += sum_e structure*qs + q.bsk ==============
// qs layout: [(b*S+q)*H + h][DEP]  (natural output of the qs GEMM)
__global__ void __launch_bounds__(256) struct_score_kernel(const float* qs, const float* Q,
                                                           const float* bsk, const float* st,
                                                           float* scores) {
  const int q = blockIdx.x, b = blockIdx.y, t = threadIdx.x;
  __shared__ float qsL[cH * cDEP];
  __shared__ float qbL[cH];
  for (int i = t; i < cH * cDEP; i += 256) {
    int hh = i / cDEP, e = i % cDEP;
    qsL[i] = qs[((long long)(b * cS + q) * cH + hh) * cDEP + e];
  }
  if (t < cH) {
    float acc = 0.f;
    const float* qp = Q + (long long)(b * cS + q) * cD + t * cDH;
    for (int d = 0; d < cDH; ++d) acc += qp[d] * bsk[d];
    qbL[t] = acc;
  }
  __syncthreads();
  const int k = t;
  const float* sp = st + ((long long)(b * cS + q) * cS + k) * cDEP;
  float acc[cH];
#pragma unroll
  for (int hh = 0; hh < cH; ++hh) acc[hh] = 0.f;
  for (int e = 0; e < cDEP; ++e) {
    float sv = sp[e];
#pragma unroll
    for (int hh = 0; hh < cH; ++hh) acc[hh] += sv * qsL[hh * cDEP + e];
  }
#pragma unroll
  for (int hh = 0; hh < cH; ++hh) {
    long long idx = ((long long)(b * cH + hh) * cS + q) * cS + k;
    scores[idx] += acc[hh] + qbL[hh];
  }
}

// ============== mask + row softmax (in place) ==============
__global__ void __launch_bounds__(256) mask_softmax_kernel(float* scores, const int* adj,
                                                           const int* lengths) {
  const int blk = blockIdx.x;               // (b*H+h)*S + q
  const int q = blk & (cS - 1);
  const int b = blk / (cH * cS);
  const int k = threadIdx.x;
  long long idx = (long long)blk * cS + k;
  float v = scores[idx];
  if (adj[(long long)(b * cS + q) * cS + k] == 0) v = NEGV;
  if (k >= lengths[b]) v = NEGV;
  __shared__ float red[256];
  red[k] = v; __syncthreads();
  for (int s = 128; s > 0; s >>= 1) { if (k < s) red[k] = fmaxf(red[k], red[k + s]); __syncthreads(); }
  float mx = red[0]; __syncthreads();
  float e = __expf(v - mx);
  red[k] = e; __syncthreads();
  for (int s = 128; s > 0; s >>= 1) { if (k < s) red[k] += red[k + s]; __syncthreads(); }
  scores[idx] = e / red[0];
}

// ============== as[b,h,q,e] = sum_k attn * structure ==============
__global__ void __launch_bounds__(512) struct_attn_kernel(const float* attn, const float* st,
                                                          float* as_out) {
  const int q = blockIdx.x, b = blockIdx.y, t = threadIdx.x;
  __shared__ float attnL[cH * cS];
  for (int i = t; i < cH * cS; i += 512) {
    int hh = i >> 8, k = i & 255;
    attnL[i] = attn[((long long)(b * cH + hh) * cS + q) * cS + k];
  }
  __syncthreads();
  for (int o = t; o < cH * cDEP; o += 512) {
    int hh = o / cDEP, e = o % cDEP;
    const float* sp = st + ((long long)(b * cS + q) * cS) * cDEP + e;
    float acc = 0.f;
    for (int k = 0; k < cS; ++k) acc += attnL[hh * cS + k] * sp[(long long)k * cDEP];
    as_out[((long long)(b * cH + hh) * cS + q) * cDEP + e] = acc;
  }
}

// ============== LayerNorm over 768, optional output stride ==============
__global__ void __launch_bounds__(256) ln_kernel(const float* in, float* out,
                                                 const float* g, const float* bb, int ldo) {
  const int row = blockIdx.x, t = threadIdx.x;
  const float* x = in + (long long)row * cD;
  float v0 = x[t], v1 = x[t + 256], v2 = x[t + 512];
  __shared__ float red[256];
  red[t] = v0 + v1 + v2; __syncthreads();
  for (int s = 128; s > 0; s >>= 1) { if (t < s) red[t] += red[t + s]; __syncthreads(); }
  float mu = red[0] / (float)cD; __syncthreads();
  float d0 = v0 - mu, d1 = v1 - mu, d2 = v2 - mu;
  red[t] = d0 * d0 + d1 * d1 + d2 * d2; __syncthreads();
  for (int s = 128; s > 0; s >>= 1) { if (t < s) red[t] += red[t + s]; __syncthreads(); }
  float inv = rsqrtf(red[0] / (float)cD + 1e-6f);
  float* o = out + (long long)row * ldo;
  o[t]       = d0 * inv * g[t]       + bb[t];
  o[t + 256] = d1 * inv * g[t + 256] + bb[t + 256];
  o[t + 512] = d2 * inv * g[t + 512] + bb[t + 512];
}

// ============== build val = [h_token | pos_cur | 0pad] ==============
__global__ void __launch_bounds__(256) build_val_kernel(const float* h, const float* cur, float* val) {
  long long i = (long long)blockIdx.x * 256 + threadIdx.x;
  const long long total = (long long)cB * cS * cVALF;
  if (i >= total) return;
  int f = (int)(i % cVALF);
  long long bs = i / cVALF;
  float v;
  if (f < cD)           v = h[bs * cD + f];
  else if (f < cD+cPOS) v = cur[bs * cPOS + (f - cD)];
  else                  v = 0.f;
  val[i] = v;
}

// ============== faithful padded-window softmax + apply ==============
__device__ __forceinline__ float local_p(const float* row, int q, int j) {
  bool win = (j >= q) && (j < q + cWSZ);
  if (!win) return 0.f;                                  // attn_p * 0
  if (j < cWSZ || j >= cS + cWSZ) return NEGV;           // NEG pad inside window
  return row[j - cWSZ];
}

__global__ void __launch_bounds__(256) local_attn_kernel(const float* sc, const float* cur,
                                                         float* nxt) {
  const int q = blockIdx.x, b = blockIdx.y, t = threadIdx.x;
  const float* row = sc + ((long long)b * cS + q) * cS;
  __shared__ float red[256];
  __shared__ float wL[cS];
  const int total = cS + 2 * cWSZ;                       // 262
  float p0 = local_p(row, q, t);
  bool has2 = (t + 256) < total;
  float p1 = has2 ? local_p(row, q, t + 256) : -3.4e38f;
  red[t] = fmaxf(p0, p1); __syncthreads();
  for (int s = 128; s > 0; s >>= 1) { if (t < s) red[t] = fmaxf(red[t], red[t + s]); __syncthreads(); }
  float mx = red[0]; __syncthreads();
  red[t] = __expf(p0 - mx) + (has2 ? __expf(p1 - mx) : 0.f); __syncthreads();
  for (int s = 128; s > 0; s >>= 1) { if (t < s) red[t] += red[t + s]; __syncthreads(); }
  float sum = red[0];
  wL[t] = __expf(local_p(row, q, t + cWSZ) - mx) / sum;  // slice [WSZ : WSZ+S]
  __syncthreads();
  if (t < cPOS) {
    float acc = 0.f;
    for (int k = 0; k < cS; ++k) acc += wL[k] * cur[((long long)b * cS + k) * cPOS + t];
    nxt[((long long)b * cS + q) * cPOS + t] = acc;
  }
}

// ============== packing kernels for cat = [tgat | pos | h_token | pad] ==============
__global__ void __launch_bounds__(256) pack_h_kernel(const float* h, float* cat) {
  long long i = (long long)blockIdx.x * 256 + threadIdx.x;
  const long long total = (long long)cB * cS * cD;
  if (i >= total) return;
  int c = (int)(i % cD); long long bs = i / cD;
  cat[bs * cCATF + (cD + cPOS) + c] = h[i];
}
__global__ void __launch_bounds__(256) pack_pos_kernel(const float* p, float* cat) {
  long long i = (long long)blockIdx.x * 256 + threadIdx.x;
  const long long total = (long long)cB * cS * cPOS;
  if (i >= total) return;
  int c = (int)(i % cPOS); long long bs = i / cPOS;
  cat[bs * cCATF + cD + c] = p[i];
}

// ================= host-side orchestration =================
static void launch_gemm(hipStream_t stream, const float* A, const float* Bm, const _Float16* B16,
                        const float* bias, const float* resid, float* C, int M, int N, int K,
                        int lda, int ldbK, int ldbN, int ldb16, int ldc,
                        long long sAz, long long sBz, long long sCz,
                        int mode, int batch, int flags, float scale) {
  GemmP g{A, Bm, B16, bias, resid, C, M, N, K, lda, ldbK, ldbN, ldb16, ldc,
          sAz, sBz, sCz, mode, flags, scale};
  dim3 grid((M + 63) / 64, (N + 63) / 64, batch);
  wmma_gemm_kernel<<<grid, dim3(256), 0, stream>>>(g);
}

static void launch_conv_wt(hipStream_t stream, const float* in, _Float16* out,
                           int K, int N, int sK, int sN, int Kpad, int Npad) {
  long long total = (long long)Npad * Kpad;
  conv_wt_kernel<<<(int)((total + 255) / 256), 256, 0, stream>>>(in, out, K, N, sK, sN, Kpad, total);
}

extern "C" void kernel_launch(void* const* d_in, const int* in_sizes, int n_in,
                              void* d_out, int out_size, void* d_ws, size_t ws_size,
                              hipStream_t stream) {
  const float* h_token = (const float*)d_in[0];
  const int*   adj     = (const int*)  d_in[1];
  const float* dep     = (const float*)d_in[2];
  const int*   lengths = (const int*)  d_in[3];
  const float* pos_e   = (const float*)d_in[4];
  const float* Wq  = (const float*)d_in[5];  const float* bq  = (const float*)d_in[6];
  const float* Wk  = (const float*)d_in[7];  const float* bk  = (const float*)d_in[8];
  const float* Wv  = (const float*)d_in[9];  const float* bv  = (const float*)d_in[10];
  const float* Wo  = (const float*)d_in[11]; const float* bo  = (const float*)d_in[12];
  const float* Wsk = (const float*)d_in[13]; const float* bsk = (const float*)d_in[14];
  const float* Wsv = (const float*)d_in[15]; const float* bsv = (const float*)d_in[16];
  const float* lng = (const float*)d_in[17]; const float* lnb = (const float*)d_in[18];
  const float* W1  = (const float*)d_in[19]; const float* b1  = (const float*)d_in[20];
  const float* W2  = (const float*)d_in[21]; const float* b2  = (const float*)d_in[22];
  const float* fg  = (const float*)d_in[23]; const float* fb  = (const float*)d_in[24];
  const float* fcw = (const float*)d_in[25]; const float* fcb = (const float*)d_in[26];
  float* out = (float*)d_out;
  (void)in_sizes; (void)n_in; (void)out_size; (void)ws_size;

  const long long nBS_D = (long long)cB * cS * cD;           // 3,145,728
  float* p = (float*)d_ws;
  size_t o = 0;
  auto take = [&](size_t n) { float* r = p + o; o += n; return r; };
  float* x    = take(nBS_D);                 // activations
  float* xa   = take(nBS_D);                 // post-attn residual
  float* Qb   = take(nBS_D);
  float* Kb   = take(nBS_D);
  float* Vb   = take(nBS_D);
  float* sc   = take((size_t)cB * cH * cS * cS);   // 12,582,912
  float* qsb  = take((size_t)cB * cS * cH * cDEP); // 1,966,080
  float* asb  = take((size_t)cB * cH * cS * cDEP); // 1,966,080
  float* valb = take((size_t)cB * cS * cVALF);     // 3,276,800
  float* posA = take((size_t)cB * cS * cPOS);
  float* posB = take((size_t)cB * cS * cPOS);
  float* cat  = take((size_t)cB * cS * cCATF);     // 6,422,528
  // ---- f16 regions (32B-aligned: all offsets are multiples of 256B) ----
  _Float16* ph = (_Float16*)(p + o);
  size_t oh = 0;
  auto takeh = [&](size_t n) { _Float16* r = ph + oh; oh += n; return r; };
  _Float16* wqT[cL]; _Float16* wkT[cL]; _Float16* wvT[cL]; _Float16* woT[cL];
  _Float16* w1T[cL]; _Float16* w2T[cL]; _Float16* wskT[cL]; _Float16* wsvT[cL];
  for (int i = 0; i < cL; ++i) {
    wqT[i] = takeh((size_t)cD * cD);   wkT[i] = takeh((size_t)cD * cD);
    wvT[i] = takeh((size_t)cD * cD);   woT[i] = takeh((size_t)cD * cD);
    w1T[i] = takeh((size_t)cDFF * cD); w2T[i] = takeh((size_t)cD * cDFF);
    wskT[i] = takeh((size_t)64 * 64);  wsvT[i] = takeh((size_t)64 * 64);
  }
  _Float16* fcT  = takeh((size_t)cDFF * cCATF);            // [768][1568]
  _Float16* K16  = takeh((size_t)nBS_D);                   // f16 copy of K
  _Float16* VT16 = takeh((size_t)cB * cH * cDH * cS);      // per-head transposed V
  float* hb    = Qb;   // LN output reuses Q (dead by then)
  float* inter = Vb;   // FFN intermediate reuses V
  float* ctx   = Kb;   // ctx reuses K

  // ---- one-time weight conversion to transposed/padded f16 ----
  for (int i = 0; i < cL; ++i) {
    const long long wo_ = (long long)i * cD * cD;
    launch_conv_wt(stream, Wq + wo_, wqT[i], cD, cD, cD, 1, cD, cD);
    launch_conv_wt(stream, Wk + wo_, wkT[i], cD, cD, cD, 1, cD, cD);
    launch_conv_wt(stream, Wv + wo_, wvT[i], cD, cD, cD, 1, cD, cD);
    launch_conv_wt(stream, Wo + wo_, woT[i], cD, cD, cD, 1, cD, cD);
    launch_conv_wt(stream, W1 + (long long)i * cD * cDFF, w1T[i], cD, cDFF, cDFF, 1, cD, cDFF);
    launch_conv_wt(stream, W2 + (long long)i * cDFF * cD, w2T[i], cDFF, cD, cD, 1, cDFF, cD);
    // qs GEMM B(k=dh, n=e) = Wsk[n*DH+k]  -> sK=1, sN=DH ; pad N 40->64, Kpad=64
    launch_conv_wt(stream, Wsk + (long long)i * cDEP * cDH, wskT[i], cDH, cDEP, 1, cDH, 64, 64);
    // sv GEMM B(k=e, n=dh) = Wsv[k*DH+n]  -> sK=DH, sN=1 ; pad K 40->64
    launch_conv_wt(stream, Wsv + (long long)i * cDEP * cDH, wsvT[i], cDEP, cDH, cDH, 1, 64, 64);
  }
  launch_conv_wt(stream, fcw, fcT, 2 * cD + cPOS, cDFF, cDFF, 1, cCATF, cDFF);

  hipMemcpyAsync(x, h_token, nBS_D * sizeof(float), hipMemcpyDeviceToDevice, stream);

  const int M = cB * cS;     // 4096
  for (int i = 0; i < cL; ++i) {
    const float* bqi = bq + (long long)i * cD;  const float* bki = bk + (long long)i * cD;
    const float* bvi = bv + (long long)i * cD;  const float* boi = bo + (long long)i * cD;
    const float* bski = bsk + (long long)i * cDH; const float* bsvi = bsv + (long long)i * cDH;
    const float* b1i = b1 + (long long)i * cDFF;  const float* b2i = b2 + (long long)i * cD;

    // Q (pre-scaled by 1/sqrt(DH)), K, V projections  (f16 transposed weights)
    launch_gemm(stream, x, nullptr, wqT[i], bqi, nullptr, Qb, M, cD, cD,
                cD, 0, 0, cD, cD, 0,0,0, 0, 1, 0, 0.125f);
    launch_gemm(stream, x, nullptr, wkT[i], bki, nullptr, Kb, M, cD, cD,
                cD, 0, 0, cD, cD, 0,0,0, 0, 1, 0, 1.f);
    launch_gemm(stream, x, nullptr, wvT[i], bvi, nullptr, Vb, M, cD, cD,
                cD, 0, 0, cD, cD, 0,0,0, 0, 1, 0, 1.f);
    // f16 copies for attention B operands
    conv_cast_kernel<<<(int)((nBS_D + 255) / 256), 256, 0, stream>>>(Kb, K16, nBS_D);
    conv_vt_kernel<<<(int)((nBS_D + 255) / 256), 256, 0, stream>>>(Vb, VT16);
    // qs = Q_heads @ Wsk^T : [B*S*H,64]@[64,40]
    launch_gemm(stream, Qb, nullptr, wskT[i], nullptr, nullptr, qsb, M * cH, cDEP, cDH,
                cDH, 0, 0, 64, cDEP, 0,0,0, 0, 1, 0, 1.f);
    // scores = Q @ K^T per (b,h)   (B16 = K16, column stride = D)
    launch_gemm(stream, Qb, nullptr, K16, nullptr, nullptr, sc, cS, cS, cDH,
                cD, 0, 0, cD, cS, 0,0,0, 1, cB * cH, 0, 1.f);
    struct_score_kernel<<<dim3(cS, cB), 256, 0, stream>>>(qsb, Qb, bski, dep, sc);
    mask_softmax_kernel<<<cB * cH * cS, 256, 0, stream>>>(sc, adj, lengths);
    struct_attn_kernel<<<dim3(cS, cB), 512, 0, stream>>>(sc, dep, asb);
    // ctx = attn @ V per (b,h)  (B16 = VT16 [z][d][k], ldb16 = S)
    launch_gemm(stream, sc, nullptr, VT16, nullptr, nullptr, ctx, cS, cDH, cS,
                cS, 0, 0, cS, cD, 0,0,0, 2, cB * cH, 0, 1.f);
    // ctx += as @ Wsv + bsv   (beta=1; sum_k attn == 1 folds the bias exactly)
    launch_gemm(stream, asb, nullptr, wsvT[i], bsvi, nullptr, ctx, cS, cDH, cDEP,
                cDEP, 0, 0, 64, cD, 0,0,0, 3, cB * cH, GF_ACC, 1.f);
    // out-projection + residual
    launch_gemm(stream, ctx, nullptr, woT[i], boi, x, xa, M, cD, cD,
                cD, 0, 0, cD, cD, 0,0,0, 0, 1, 0, 1.f);
    // FFN: LN -> relu(W1) -> W2 + residual
    ln_kernel<<<M, 256, 0, stream>>>(xa, hb, lng + (long long)i * cD, lnb + (long long)i * cD, cD);
    launch_gemm(stream, hb, nullptr, w1T[i], b1i, nullptr, inter, M, cDFF, cD,
                cD, 0, 0, cD, cDFF, 0,0,0, 0, 1, GF_RELU, 1.f);
    launch_gemm(stream, inter, nullptr, w2T[i], b2i, xa, x, M, cD, cDFF,
                cDFF, 0, 0, cDFF, cD, 0,0,0, 0, 1, 0, 1.f);
  }

  // final LN straight into cat cols [0,768)
  ln_kernel<<<M, 256, 0, stream>>>(x, cat, fg, fb, cCATF);
  pack_h_kernel<<<(int)((nBS_D + 255) / 256), 256, 0, stream>>>(h_token, cat);

  // local position attention (faithful padded-mask softmax), NLOC iterations
  const float* cur = pos_e;
  float* nxt = posA;
  const long long nval = (long long)cB * cS * cVALF;
  for (int it = 0; it < cNLOC; ++it) {
    build_val_kernel<<<(int)((nval + 255) / 256), 256, 0, stream>>>(h_token, cur, valb);
    launch_gemm(stream, valb, valb, nullptr, nullptr, nullptr, sc, cS, cS, cD + cPOS,
                cVALF, 1, cVALF, 0, cS,
                (long long)cS * cVALF, (long long)cS * cVALF, (long long)cS * cS,
                0, cB, 0, 1.f);
    local_attn_kernel<<<dim3(cS, cB), 256, 0, stream>>>(sc, cur, nxt);
    cur = nxt;
    nxt = (it == 0) ? posB : posA;
  }
  const long long npos = (long long)cB * cS * cPOS;
  pack_pos_kernel<<<(int)((npos + 255) / 256), 256, 0, stream>>>(cur, cat);

  // final FC: [4096,1566] @ [1566(pad 1568),768] + fc_b  -> d_out
  launch_gemm(stream, cat, nullptr, fcT, fcb, nullptr, out, M, cDFF, 2 * cD + cPOS,
              cCATF, 0, 0, cCATF, cDFF, 0,0,0, 0, 1, 0, 1.f);
}
